// SimilarityLSTM_88098369176043
// MI455X (gfx1250) — matrix operand order repaired and checked by reading, not compiled, not run on hardware
//
#include <hip/hip_runtime.h>
#include <hip/hip_bf16.h>

// ---------------------------------------------------------------------------
// SimilarityLSTM on MI455X (gfx1250, wave32, WMMA)
//
// Phase 1: gather + bf16 conversion (bandwidth trivial)
// Phase 2: xp = x @ W_ih^T + (b_ih+b_hh) as bf16 WMMA GEMM, f32 accum
// Phase 3: sequential LSTM recurrence, W_hh resident in LDS across 16 WGs
//          per side (async-loaded to LDS via GLOBAL_LOAD_ASYNC_TO_LDS_B128),
//          per-step h exchange via L2 + atomic split-barrier
// Phase 4: exp(-sum|hl-hr|) reduction
// ---------------------------------------------------------------------------

#define SEQ    2048
#define IN_DIM 256
#define H      512
#define H4     2048   // 4*H

typedef __attribute__((ext_vector_type(16))) __bf16 v16bf;
typedef __attribute__((ext_vector_type(8)))  float  v8f;
typedef __attribute__((ext_vector_type(4)))  int    v4i;
typedef __attribute__((address_space(1))) v4i* v4i_gptr;   // global
typedef __attribute__((address_space(3))) v4i* v4i_lptr;   // LDS

__device__ __forceinline__ unsigned short f2bf(float f) {
    unsigned u = __float_as_uint(f);
    unsigned r = (u + 0x7FFFu + ((u >> 16) & 1u)) >> 16;  // RNE
    return (unsigned short)r;
}
__device__ __forceinline__ float bfhi2f(unsigned u) {           // bf16 in [31:16]
    return __uint_as_float(u & 0xFFFF0000u);
}
__device__ __forceinline__ float bflo2f(unsigned u) {           // bf16 in [15:0]
    return __uint_as_float(u << 16);
}
__device__ __forceinline__ float sigf(float x) { return 1.0f / (1.0f + __expf(-x)); }

// ---------------- Phase 1: gather + convert ----------------
__global__ void gather_embed_kernel(const int* __restrict__ lin,
                                    const int* __restrict__ rin,
                                    const float* __restrict__ emb,
                                    unsigned short* __restrict__ xl,
                                    unsigned short* __restrict__ xr) {
    int tid = blockIdx.x * blockDim.x + threadIdx.x;   // SEQ*IN_DIM
    if (tid >= SEQ * IN_DIM) return;
    int s = tid >> 8, d = tid & 255;
    xl[tid] = f2bf(emb[(size_t)lin[s] * IN_DIM + d]);
    xr[tid] = f2bf(emb[(size_t)rin[s] * IN_DIM + d]);
}

__global__ void convert_bf16_kernel(const float* __restrict__ src,
                                    unsigned short* __restrict__ dst, int n) {
    int tid = blockIdx.x * blockDim.x + threadIdx.x;
    if (tid < n) dst[tid] = f2bf(src[tid]);
}

__global__ void bias_sum_kernel(const float* __restrict__ b_ih,
                                const float* __restrict__ b_hh,
                                float* __restrict__ bias) {
    int tid = blockIdx.x * blockDim.x + threadIdx.x;
    if (tid < H4) bias[tid] = b_ih[tid] + b_hh[tid];
}

// ---------------- Phase 2: xp = x @ W_ih^T + b  (WMMA bf16) ----------------
// grid = (SEQ/16, H4/128, 2 sides), block = 256 (8 waves).
// Wave w computes the 16x16 tile at (m0 = bx*16, n0 = by*128 + w*16).
__global__ void gemm_xp_kernel(const unsigned short* __restrict__ xl,
                               const unsigned short* __restrict__ xr,
                               const unsigned short* __restrict__ wih,
                               const float* __restrict__ bias,
                               float* __restrict__ xpl,
                               float* __restrict__ xpr) {
    const int side = blockIdx.z;
    const unsigned short* X  = side ? xr  : xl;
    float*                XP = side ? xpr : xpl;

    const int wave = threadIdx.x >> 5;
    const int lane = threadIdx.x & 31;
    const int lr   = lane & 15;      // M (A) / N (B) index within tile
    const int hi   = lane >> 4;      // lane-half selects K sub-blocks

    const int m0 = blockIdx.x * 16;
    const int n0 = blockIdx.y * 128 + wave * 16;

    const unsigned short* arow = X   + (size_t)(m0 + lr) * IN_DIM;  // A row-major
    const unsigned short* brow = wih + (size_t)(n0 + lr) * IN_DIM;  // B[k][n] = W[n][k]

    v8f acc = {0.f, 0.f, 0.f, 0.f, 0.f, 0.f, 0.f, 0.f};

    union Frag { uint4 q[2]; v16bf v; };

#pragma unroll
    for (int k0 = 0; k0 < IN_DIM; k0 += 32) {
        // A 16x32 bf16 layout (ISA 7.12.2): lanes 0-15 K=0..7 & 16..23,
        // lanes 16-31 K=8..15 & 24..31
        Frag fa, fb;
        fa.q[0] = *(const uint4*)(arow + k0 + hi * 8);
        fa.q[1] = *(const uint4*)(arow + k0 + 16 + hi * 8);
        // B 32x16 bf16: same K order as A (lanes 0-15 K=0..7 & 16..23, lanes 16-31 K=8..15 & 24..31)
        fb.q[0] = *(const uint4*)(brow + k0 + hi * 8);
        fb.q[1] = *(const uint4*)(brow + k0 + 16 + hi * 8);

        if (k0 + 32 < IN_DIM) {   // global_prefetch_b8 next K tile
            __builtin_prefetch(arow + k0 + 32, 0, 1);
            __builtin_prefetch(brow + k0 + 32, 0, 1);
        }

        acc = __builtin_amdgcn_wmma_f32_16x16x32_bf16(
            /*neg_a=*/false, fa.v, /*neg_b=*/false, fb.v,
            /*c_mod=*/(short)0, acc, /*reuse_a=*/false, /*reuse_b=*/false);
    }

    const int   n  = n0 + lr;
    const float bn = bias[n];
#pragma unroll
    for (int r = 0; r < 8; ++r) {
        // C/D layout: VGPR r -> M = r (lanes 0-15) or r+8 (lanes 16-31)
        XP[(size_t)(m0 + hi * 8 + r) * H4 + n] = acc[r] + bn;
    }
}

// ---------------- Phase 3: sequential recurrence ----------------
// 16 WGs per side; WG `wg` owns hidden slice j0 = wg*32 and all 4 gates for
// it: 128 rows x 512 cols of W_hh, bf16-resident in (dynamic) LDS = 128 KB,
// moved in via GLOBAL_LOAD_ASYNC_TO_LDS_B128 (ASYNCcnt) when available.
// Per step: broadcast h (512 f32, double-buffered in L2), 128 dot-512s,
// gates, write own 32-wide h slice, device-scope split-barrier.
#define NWG_SIDE 16
#define REC_THREADS 128

__global__ void lstm_rec_kernel(const float* __restrict__ xpl,
                                const float* __restrict__ xpr,
                                const unsigned short* __restrict__ whh,
                                float* __restrict__ hstate,   // [2 side][2 buf][512]
                                int* __restrict__ counters) { // [2 side][64-pad]
    extern __shared__ char smem[];
    unsigned short* Wlds = (unsigned short*)smem;                       // 128*512 bf16
    float* hbuf = (float*)(smem + REC_THREADS * H * sizeof(unsigned short)); // 512 f32
    float* gbuf = hbuf + H;                                             // 128 f32

    const int side = blockIdx.x >> 4;
    const int wg   = blockIdx.x & (NWG_SIDE - 1);
    const int tid  = threadIdx.x;
    const int j0   = wg * 32;
    const int g    = tid >> 5;                 // gate 0..3 (i,f,g,o)
    const int jj   = tid & 31;
    const int row  = g * H + j0 + jj;          // global gate row in [0,2048)

    const float* XP = side ? xpr : xpl;
    float* hs       = hstate + side * (2 * H);
    int*   cnt      = counters + side * 64;

    // Preload this WG's 4 contiguous 32-row chunks of W_hh (bf16) into LDS.
    // Preferred path: async global->LDS copy (no VGPR staging, ASYNCcnt).
    {
        const uint4* src = (const uint4*)whh;        // 64 uint4 per 512-col row
        uint4*       dst = (uint4*)Wlds;
        const int total = REC_THREADS * H / 8;       // 8192 uint4 = 128 KB
#if __has_builtin(__builtin_amdgcn_global_load_async_to_lds_b128)
        for (int q = tid; q < total; q += REC_THREADS) {
            int chunk = q >> 11;                     // 2048 uint4 per 32-row chunk
            int off   = q & 2047;
            const uint4* gp = src + (size_t)(chunk * H + j0) * 64 + off;
            __builtin_amdgcn_global_load_async_to_lds_b128(
                (v4i_gptr)gp,           // global src (AS1 int4*, const stripped)
                (v4i_lptr)(dst + q),    // LDS dst (AS3 int4*)
                /*imm offset=*/0, /*cpol=*/0);
        }
#if __has_builtin(__builtin_amdgcn_s_wait_asynccnt)
        __builtin_amdgcn_s_wait_asynccnt(0);
#else
        asm volatile("s_wait_asynccnt 0x0" ::: "memory");
#endif
#else
        for (int q = tid; q < total; q += REC_THREADS) {
            int chunk = q >> 11;
            int off   = q & 2047;
            dst[q] = src[(size_t)(chunk * H + j0) * 64 + off];
        }
#endif
    }
    __syncthreads();

    float c_reg = 0.0f;   // valid for tid < 32 (hidden unit j0+tid)

    for (int t = 0; t < SEQ; ++t) {
        const int p  = t & 1;          // read buffer
        const int pn = p ^ 1;          // write buffer

        // h[t] -> LDS (zero-initialized buffer 0 covers t == 0)
        for (int i = tid; i < H; i += REC_THREADS) hbuf[i] = hs[p * H + i];
        __syncthreads();

        // gate pre-activation: xp[t][row] + dot(W_hh[row], h)
        float acc = XP[(size_t)t * H4 + row];
        const unsigned* wq = (const unsigned*)(Wlds + tid * H);
#pragma unroll 8
        for (int k2 = 0; k2 < H / 2; ++k2) {
            unsigned w = wq[k2];
            acc = fmaf(bflo2f(w), hbuf[2 * k2],     acc);
            acc = fmaf(bfhi2f(w), hbuf[2 * k2 + 1], acc);
        }
        gbuf[tid] = acc;
        __syncthreads();

        if (tid < 32) {
            float gi = sigf(gbuf[tid]);
            float gf = sigf(gbuf[32 + tid]);
            float gg = tanhf(gbuf[64 + tid]);
            float go = sigf(gbuf[96 + tid]);
            c_reg = gf * c_reg + gi * gg;
            hs[pn * H + j0 + tid] = go * tanhf(c_reg);
        }
        __threadfence();       // release h slice to device scope
        __syncthreads();

        // device-scope split-barrier across the 16 WGs of this side
        if (tid == 0) {
            __hip_atomic_fetch_add(cnt, 1, __ATOMIC_ACQ_REL, __HIP_MEMORY_SCOPE_AGENT);
            while (__hip_atomic_load(cnt, __ATOMIC_ACQUIRE, __HIP_MEMORY_SCOPE_AGENT)
                   < NWG_SIDE * (t + 1)) { }
        }
        __syncthreads();
    }
    // After t = SEQ-1 (odd), final h lives in buffer 0 of each side.
}

// ---------------- Phase 4: exp(-sum|hl-hr|) ----------------
__global__ void reduce_kernel(const float* __restrict__ hstate, float* __restrict__ out) {
    __shared__ float red[512];
    int tid = threadIdx.x;
    const float* hl = hstate;              // side 0, buffer 0
    const float* hr = hstate + 2 * H;      // side 1, buffer 0
    red[tid] = fabsf(hl[tid] - hr[tid]);
    __syncthreads();
#pragma unroll
    for (int s = 256; s > 0; s >>= 1) {
        if (tid < s) red[tid] += red[tid + s];
        __syncthreads();
    }
    if (tid == 0) out[0] = __expf(-red[0]);
}

// ---------------- launcher ----------------
extern "C" void kernel_launch(void* const* d_in, const int* in_sizes, int n_in,
                              void* d_out, int out_size, void* d_ws, size_t ws_size,
                              hipStream_t stream) {
    const int*   lin  = (const int*)d_in[0];
    const int*   rin  = (const int*)d_in[1];
    const float* emb  = (const float*)d_in[2];
    const float* w_ih = (const float*)d_in[3];
    const float* w_hh = (const float*)d_in[4];
    const float* b_ih = (const float*)d_in[5];
    const float* b_hh = (const float*)d_in[6];
    float* out = (float*)d_out;

    char* ws = (char*)d_ws;
    size_t off = 0;
    unsigned short* xl   = (unsigned short*)(ws + off); off += (size_t)SEQ * IN_DIM * 2;  // 1 MB
    unsigned short* xr   = (unsigned short*)(ws + off); off += (size_t)SEQ * IN_DIM * 2;  // 1 MB
    unsigned short* wihb = (unsigned short*)(ws + off); off += (size_t)H4 * IN_DIM * 2;   // 1 MB
    unsigned short* whhb = (unsigned short*)(ws + off); off += (size_t)H4 * H * 2;        // 2 MB
    float* bias = (float*)(ws + off); off += (size_t)H4 * 4;                              // 8 KB
    float* xpl  = (float*)(ws + off); off += (size_t)SEQ * H4 * 4;                        // 16 MB
    float* xpr  = (float*)(ws + off); off += (size_t)SEQ * H4 * 4;                        // 16 MB
    size_t hoff = off;
    float* hstate = (float*)(ws + off); off += 2 * 2 * H * 4;                             // 8 KB
    int* counters = (int*)(ws + off);   off += 512;

    // zero h state (h0 = c0 = 0) and the barrier counters every launch
    (void)hipMemsetAsync(ws + hoff, 0, 2 * 2 * H * 4 + 512, stream);

    // Phase 1
    gather_embed_kernel<<<(SEQ * IN_DIM + 255) / 256, 256, 0, stream>>>(lin, rin, emb, xl, xr);
    convert_bf16_kernel<<<(H4 * IN_DIM + 255) / 256, 256, 0, stream>>>(w_ih, wihb, H4 * IN_DIM);
    convert_bf16_kernel<<<(H4 * H + 255) / 256, 256, 0, stream>>>(w_hh, whhb, H4 * H);
    bias_sum_kernel<<<(H4 + 255) / 256, 256, 0, stream>>>(b_ih, b_hh, bias);

    // Phase 2: WMMA GEMM, both sides
    gemm_xp_kernel<<<dim3(SEQ / 16, H4 / 128, 2), 256, 0, stream>>>(xl, xr, wihb, bias, xpl, xpr);

    // Phase 3: recurrence; 128 KB W_hh tile + h + gates as dynamic LDS
    size_t lds_bytes = (size_t)REC_THREADS * H * 2 + H * 4 + REC_THREADS * 4; // 133,632 B
    lstm_rec_kernel<<<2 * NWG_SIDE, REC_THREADS, lds_bytes, stream>>>(xpl, xpr, whhb,
                                                                     hstate, counters);

    // Phase 4
    reduce_kernel<<<1, 512, 0, stream>>>(hstate, out);
}
